// Swin_base_patch4_window7_224_in22k_63256278336151
// MI455X (gfx1250) — compile-verified, matrix operand order checked by reading, not run on hardware
//
#include <hip/hip_runtime.h>
#include <math.h>

typedef __attribute__((ext_vector_type(16))) _Float16 v16h;
typedef __attribute__((ext_vector_type(8)))  _Float16 v8h;
typedef __attribute__((ext_vector_type(4)))  _Float16 v4h;
typedef __attribute__((ext_vector_type(8)))  float    v8f;
typedef __attribute__((ext_vector_type(4)))  float    v4f;

// ---------------------------------------------------------------- WMMA helpers

__device__ __forceinline__ v8f wmma32(v16h a, v16h b, v8f c) {
  return __builtin_amdgcn_wmma_f32_16x16x32_f16(
      false, a, false, b, (short)0, c, false, false);
}

// Fragment loader for K-contiguous LDS layouts (ld multiple of 8 halves ->
// two aligned ds_load_b128 per fragment).
// A-matrix: rows are M; B-matrix (N-major storage [n][k]): rows are N.
__device__ __forceinline__ v16h fragr(const _Float16* p, int ld, int row0, int k0) {
  int lane = threadIdx.x & 31;
  const _Float16* q = p + (row0 + (lane & 15)) * ld + k0 + ((lane >> 4) << 3);
  v8h lo = *(const v8h*)(q);
  v8h hi = *(const v8h*)(q + 16);
  return __builtin_shufflevector(lo, hi, 0, 1, 2, 3, 4, 5, 6, 7,
                                 8, 9, 10, 11, 12, 13, 14, 15);
}

// ---------------------------------------------------------------- fused GEMM
// Y[M,N] = act( X[M,K] * W[K,N] + bias ) + R   (act: 0=none, 1=exact GELU)
// 64x64 tile, 128 threads = 4 waves, each wave a 32x32 quadrant.
// Interior tiles take a register-prefetch pipelined path: next K-step's global
// loads are issued before the WMMA block so VMEM latency overlaps the matrix
// pipe; boundary tiles use a guarded generic path.
__global__ __launch_bounds__(128)
void gemm_kernel(const float* __restrict__ X, const float* __restrict__ Wt,
                 const float* __restrict__ bias, const float* __restrict__ Rres,
                 float* __restrict__ Y, int M, int N, int K, int act) {
  __shared__ __align__(16) _Float16 As[64 * 40];  // [m][k]
  __shared__ __align__(16) _Float16 Bs[64 * 40];  // [n][k] (N-major)
  const int ldA = 40, ldB = 40;
  const int t = threadIdx.x;
  const int bm = blockIdx.y * 64, bn = blockIdx.x * 64;
  const int wv = t >> 5, lane = t & 31;
  const int r0 = (wv >> 1) * 32, c0 = (wv & 1) * 32;

  v8f z = {};
  v8f a00 = z, a01 = z, a10 = z, a11 = z;

  const bool fast = (bm + 64 <= M) && (bn + 64 <= N) &&
                    ((K & 31) == 0) && ((N & 3) == 0);

  if (fast) {
    // per-thread stationary indices
    const int ra = (t + 0) >> 3, ca = ((t + 0) & 7) << 2;     // A: +i*16 rows
    const int rb = (t + 0) >> 4, cb = ((t + 0) & 15) << 2;    // B: +i*8 rows
    v4f ax[4], bx[4];
#pragma unroll
    for (int i = 0; i < 4; ++i) {
      ax[i] = *(const v4f*)(X + (size_t)(bm + ra + i * 16) * K + ca);
      bx[i] = *(const v4f*)(Wt + (size_t)(rb + i * 8) * N + bn + cb);
    }
    for (int k0 = 0; k0 < K; k0 += 32) {
      // commit current regs to LDS (f32 -> f16)
#pragma unroll
      for (int i = 0; i < 4; ++i) {
        v4h h;
#pragma unroll
        for (int q = 0; q < 4; ++q) h[q] = (_Float16)ax[i][q];
        *(v4h*)(As + (ra + i * 16) * ldA + ca) = h;
#pragma unroll
        for (int q = 0; q < 4; ++q)
          Bs[(cb + q) * ldB + rb + i * 8] = (_Float16)bx[i][q];
      }
      __syncthreads();
      // issue next tile's global loads before compute (latency overlap)
      if (k0 + 32 < K) {
#pragma unroll
        for (int i = 0; i < 4; ++i) {
          ax[i] = *(const v4f*)(X + (size_t)(bm + ra + i * 16) * K + k0 + 32 + ca);
          bx[i] = *(const v4f*)(Wt + (size_t)(k0 + 32 + rb + i * 8) * N + bn + cb);
        }
      }
      v16h fa0 = fragr(As, ldA, r0, 0);
      v16h fa1 = fragr(As, ldA, r0 + 16, 0);
      v16h fb0 = fragr(Bs, ldB, c0, 0);
      v16h fb1 = fragr(Bs, ldB, c0 + 16, 0);
      a00 = wmma32(fa0, fb0, a00);
      a01 = wmma32(fa0, fb1, a01);
      a10 = wmma32(fa1, fb0, a10);
      a11 = wmma32(fa1, fb1, a11);
      __syncthreads();
    }
  } else {
    for (int k0 = 0; k0 < K; k0 += 32) {
      for (int e = t; e < 64 * 32; e += 128) {
        int r = e >> 5, c = e & 31;
        int gm = bm + r, gk = k0 + c;
        float v = (gm < M && gk < K) ? X[(size_t)gm * K + gk] : 0.f;
        As[r * ldA + c] = (_Float16)v;
      }
      for (int e = t; e < 32 * 64; e += 128) {
        int r = e >> 6, c = e & 63;
        int gk = k0 + r, gn = bn + c;
        float v = (gk < K && gn < N) ? Wt[(size_t)gk * N + gn] : 0.f;
        Bs[c * ldB + r] = (_Float16)v;
      }
      __syncthreads();
      v16h fa0 = fragr(As, ldA, r0, 0);
      v16h fa1 = fragr(As, ldA, r0 + 16, 0);
      v16h fb0 = fragr(Bs, ldB, c0, 0);
      v16h fb1 = fragr(Bs, ldB, c0 + 16, 0);
      a00 = wmma32(fa0, fb0, a00);
      a01 = wmma32(fa0, fb1, a01);
      a10 = wmma32(fa1, fb0, a10);
      a11 = wmma32(fa1, fb1, a11);
      __syncthreads();
    }
  }

  const int cl = lane & 15, rh8 = (lane >> 4) * 8;
  v8f accs[4] = {a00, a01, a10, a11};
#pragma unroll
  for (int ti = 0; ti < 2; ++ti)
#pragma unroll
    for (int tj = 0; tj < 2; ++tj) {
      v8f acc = accs[ti * 2 + tj];
      int gn = bn + c0 + tj * 16 + cl;
#pragma unroll
      for (int g = 0; g < 8; ++g) {
        int gm = bm + r0 + ti * 16 + rh8 + g;
        if (gm < M && gn < N) {
          float v = acc[g];
          if (bias) v += bias[gn];
          if (act) v = 0.5f * v * (1.f + erff(v * 0.70710678118654752f));
          if (Rres) v += Rres[(size_t)gm * N + gn];
          Y[(size_t)gm * N + gn] = v;
        }
      }
    }
}

// ---------------------------------------------------------------- attention
__device__ __forceinline__ int regid(int h, int H) {
  return h < H - 7 ? 0 : (h < H - 3 ? 1 : 2);
}

// One block per (window-batch, head).  QKV: [B*L, 3C] in (b,l) layout.
// Out: [B*L, C] in (b,l) layout (cyclic-shift roll folded into gather/scatter).
__global__ __launch_bounds__(128)
void attn_kernel(const float* __restrict__ QKV, const float* __restrict__ rpb,
                 float* __restrict__ Out, int H, int C, int heads, int shift,
                 int nWps) {
  __shared__ __align__(16) _Float16 qh[64 * 40];  // q  [n][d]
  __shared__ __align__(16) _Float16 kh[64 * 40];  // k  [m][d]  (N-major B)
  __shared__ __align__(16) _Float16 vT[32 * 72];  // v^T[d][m]  (N-major B)
  __shared__ __align__(16) _Float16 ah[64 * 72];  // attn [n][m] (A, K-contig)
  __shared__ float af[64 * 65];

  const int t = threadIdx.x;
  const int wb = blockIdx.x, head = blockIdx.y;
  const int L = H * H;
  const int wpi = nWps * nWps;
  const int b = wb / wpi, wrem = wb % wpi;
  const int wh = wrem / nWps, ww = wrem % nWps;
  const int C3 = 3 * C;
  const int qoff = head * 32;

  auto tok_l = [&](int n) -> int {
    int i = n / 7, j = n % 7;
    int sh = wh * 7 + i + shift; if (sh >= H) sh -= H;
    int sw = ww * 7 + j + shift; if (sw >= H) sw -= H;
    return sh * H + sw;
  };

  // gather q/k/v rows (8 float4 per token), f32->f16, v transposed
  for (int e = t; e < 512; e += 128) {
    int n = e >> 3, c4 = (e & 7) << 2;
    if (n < 49) {
      size_t base = (size_t)(b * L + tok_l(n)) * C3 + qoff;
      v4f q4 = *(const v4f*)(QKV + base + c4);
      v4f k4 = *(const v4f*)(QKV + base + C + c4);
      v4f w4 = *(const v4f*)(QKV + base + 2 * C + c4);
      v4h qx, kx;
#pragma unroll
      for (int q = 0; q < 4; ++q) { qx[q] = (_Float16)q4[q]; kx[q] = (_Float16)k4[q]; }
      *(v4h*)(qh + n * 40 + c4) = qx;
      *(v4h*)(kh + n * 40 + c4) = kx;
#pragma unroll
      for (int q = 0; q < 4; ++q) vT[(c4 + q) * 72 + n] = (_Float16)w4[q];
    } else {
      v4h zh = {};
      *(v4h*)(qh + n * 40 + c4) = zh;
      *(v4h*)(kh + n * 40 + c4) = zh;
#pragma unroll
      for (int q = 0; q < 4; ++q) vT[(c4 + q) * 72 + n] = (_Float16)0.f;
    }
  }
  for (int e = t; e < 64 * 72; e += 128) ah[e] = (_Float16)0.f;
  __syncthreads();

  // --- scores = q @ k^T  (64x64 padded, K=32) ---
  const int wv = t >> 5, lane = t & 31;
  const int r0 = (wv >> 1) * 32, c0 = (wv & 1) * 32;
  v8f z = {};
  v8f s00 = z, s01 = z, s10 = z, s11 = z;
  {
    v16h fa0 = fragr(qh, 40, r0, 0);
    v16h fa1 = fragr(qh, 40, r0 + 16, 0);
    v16h fb0 = fragr(kh, 40, c0, 0);
    v16h fb1 = fragr(kh, 40, c0 + 16, 0);
    s00 = wmma32(fa0, fb0, s00);
    s01 = wmma32(fa0, fb1, s01);
    s10 = wmma32(fa1, fb0, s10);
    s11 = wmma32(fa1, fb1, s11);
  }
  const int cl = lane & 15, rh8 = (lane >> 4) * 8;
  v8f scc[4] = {s00, s01, s10, s11};
#pragma unroll
  for (int ti = 0; ti < 2; ++ti)
#pragma unroll
    for (int tj = 0; tj < 2; ++tj) {
      v8f acc = scc[ti * 2 + tj];
#pragma unroll
      for (int g = 0; g < 8; ++g)
        af[(r0 + ti * 16 + rh8 + g) * 65 + c0 + tj * 16 + cl] = acc[g];
    }
  __syncthreads();

  // --- softmax with rel-pos bias + shift mask, rows 0..48 ---
  if (t < 49) {
    const float scale = 0.17677669529663687f;  // 1/sqrt(32)
    int i1 = t / 7, j1 = t % 7;
    int lab1 = 0;
    if (shift) lab1 = regid(wh * 7 + i1, H) * 3 + regid(ww * 7 + j1, H);
    float mx = -1e30f;
    for (int m = 0; m < 49; ++m) {
      int i2 = m / 7, j2 = m % 7;
      float v = af[t * 65 + m] * scale;
      int ridx = (i1 - i2 + 6) * 13 + (j1 - j2 + 6);
      v += rpb[ridx * heads + head];
      if (shift) {
        int lab2 = regid(wh * 7 + i2, H) * 3 + regid(ww * 7 + j2, H);
        if (lab1 != lab2) v -= 100.f;
      }
      af[t * 65 + m] = v;
      mx = fmaxf(mx, v);
    }
    float s = 0.f;
    for (int m = 0; m < 49; ++m) {
      float e = __expf(af[t * 65 + m] - mx);
      af[t * 65 + m] = e;
      s += e;
    }
    float inv = 1.f / s;
    for (int m = 0; m < 49; ++m) ah[t * 72 + m] = (_Float16)(af[t * 65 + m] * inv);
  }
  __syncthreads();

  // --- out = attn @ v  (64x32, K=64): wave wv owns rows wv*16..+15 ---
  v8f o0 = z, o1 = z;
#pragma unroll
  for (int k0 = 0; k0 < 64; k0 += 32) {
    v16h fa = fragr(ah, 72, wv * 16, k0);
    v16h fb0 = fragr(vT, 72, 0, k0);
    v16h fb1 = fragr(vT, 72, 16, k0);
    o0 = wmma32(fa, fb0, o0);
    o1 = wmma32(fa, fb1, o1);
  }
#pragma unroll
  for (int g = 0; g < 8; ++g) {
    int n = wv * 16 + rh8 + g;
    if (n < 49) {
      size_t base = (size_t)(b * L + tok_l(n)) * C + qoff;
      Out[base + cl]      = o0[g];
      Out[base + 16 + cl] = o1[g];
    }
  }
}

// ---------------------------------------------------------------- LayerNorm
__global__ __launch_bounds__(128)
void ln_kernel(const float* __restrict__ X, const float* __restrict__ g,
               const float* __restrict__ bta, float* __restrict__ Y, int C) {
  __shared__ float red[128];
  const int row = blockIdx.x, t = threadIdx.x;
  const float* x = X + (size_t)row * C;
  float s = 0.f;
  for (int c = t; c < C; c += 128) s += x[c];
  red[t] = s; __syncthreads();
  for (int o = 64; o > 0; o >>= 1) { if (t < o) red[t] += red[t + o]; __syncthreads(); }
  float mu = red[0] / C; __syncthreads();
  float vv = 0.f;
  for (int c = t; c < C; c += 128) { float d = x[c] - mu; vv += d * d; }
  red[t] = vv; __syncthreads();
  for (int o = 64; o > 0; o >>= 1) { if (t < o) red[t] += red[t + o]; __syncthreads(); }
  float rstd = rsqrtf(red[0] / C + 1e-5f);
  float* y = Y + (size_t)row * C;
  for (int c = t; c < C; c += 128) y[c] = (x[c] - mu) * rstd * g[c] + bta[c];
}

// ---------------------------------------------------------------- misc
__global__ void patch_kernel(const float* __restrict__ img,
                             const float* __restrict__ Kw,
                             const float* __restrict__ bias,
                             float* __restrict__ Y) {
  int idx = blockIdx.x * blockDim.x + threadIdx.x;
  if (idx >= 8 * 56 * 56 * 128) return;
  int c = idx & 127;
  int p = idx >> 7;
  int ow = p % 56; int tmp = p / 56; int oh = tmp % 56; int b = tmp / 56;
  float acc = bias[c];
  for (int kh = 0; kh < 4; ++kh)
    for (int kw = 0; kw < 4; ++kw)
      for (int ci = 0; ci < 3; ++ci)
        acc += img[((size_t)(b * 3 + ci) * 224 + oh * 4 + kh) * 224 + ow * 4 + kw] *
               Kw[((kh * 4 + kw) * 3 + ci) * 128 + c];
  Y[(size_t)p * 128 + c] = acc;
}

__global__ void merge_kernel(const float* __restrict__ X, float* __restrict__ Y,
                             int H, int C) {
  int H2 = H / 2, L = H * H, Lo = H2 * H2;
  int idx = blockIdx.x * blockDim.x + threadIdx.x;
  if (idx >= 8 * Lo * 4 * C) return;
  int c4 = idx % (4 * C); int p = idx / (4 * C);
  int w2 = p % H2; int t2 = p / H2; int h2 = t2 % H2; int b = t2 / H2;
  int g = c4 / C, c = c4 % C;
  int dh = (g == 1 || g == 3) ? 1 : 0;
  int dw = (g >= 2) ? 1 : 0;
  Y[idx] = X[((size_t)b * L + (2 * h2 + dh) * H + (2 * w2 + dw)) * C + c];
}

__global__ void pool_kernel(const float* __restrict__ X, float* __restrict__ Y) {
  int idx = blockIdx.x * blockDim.x + threadIdx.x;
  if (idx >= 8 * 1024) return;
  int c = idx & 1023, b = idx >> 10;
  float s = 0.f;
  for (int n = 0; n < 49; ++n) s += X[((size_t)b * 49 + n) * 1024 + c];
  Y[idx] = s * (1.f / 49.f);
}

__global__ void feat_kernel(const float* __restrict__ X, float* __restrict__ F) {
  int idx = blockIdx.x * blockDim.x + threadIdx.x;
  if (idx >= 8 * 1024 * 49) return;
  int hw = idx % 49; int t = idx / 49; int c = t % 1024; int b = t / 1024;
  F[idx] = X[((size_t)b * 49 + hw) * 1024 + c];
}

// ---------------------------------------------------------------- launch
extern "C" void kernel_launch(void* const* d_in, const int* in_sizes, int n_in,
                              void* d_out, int out_size, void* d_ws, size_t ws_size,
                              hipStream_t stream) {
  (void)in_sizes; (void)n_in; (void)out_size; (void)ws_size;
  auto P = [&](int i) { return (const float*)d_in[i]; };

  float* ws  = (float*)d_ws;
  float* X   = ws;               // 3.4M floats (max B*L*C = 3.21M)
  float* LNB = ws + 3400000;     // 3.4M
  float* BIG = ws + 6800000;     // 13.2M (qkv 9.6M / mlp-hidden 12.85M)
  float* ATT = ws + 20000000;    // 3.4M (attention out; aliased as merge scratch)
  float* TMP = ATT;
  float* POOL = BIG;             // head-chain scratch aliases BIG
  float* CLS  = BIG + 10000;
  float* H1   = BIG + 200000;

  auto gemm = [&](const float* Xp, const float* Wp, const float* bp,
                  const float* Rp, float* Yp, int M, int N, int K, int act) {
    dim3 g((N + 63) / 64, (M + 63) / 64);
    gemm_kernel<<<g, dim3(128), 0, stream>>>(Xp, Wp, bp, Rp, Yp, M, N, K, act);
  };

  // patch embed + embed LN (in place)
  {
    int tot = 8 * 56 * 56 * 128;
    patch_kernel<<<(tot + 255) / 256, 256, 0, stream>>>(P(0), P(1), P(2), X);
    ln_kernel<<<8 * 3136, 128, 0, stream>>>(X, P(3), P(4), X, 128);
  }

  const int RESa[4] = {56, 28, 14, 7};
  const int DIMa[4] = {128, 256, 512, 1024};
  const int HDSa[4] = {4, 8, 16, 32};
  const int DEPa[4] = {2, 2, 18, 2};
  const int base[4] = {13, 42, 71, 308};

  for (int s = 0; s < 4; ++s) {
    int H = RESa[s], C = DIMa[s], L = H * H, M = 8 * L;
    int heads = HDSa[s], nWps = H / 7, nWB = 8 * nWps * nWps;
    for (int j = 0; j < DEPa[s]; ++j) {
      int bb = base[s] + j * 13;
      int shift = (j % 2 == 1 && H > 7) ? 3 : 0;
      ln_kernel<<<M, 128, 0, stream>>>(X, P(bb + 0), P(bb + 1), LNB, C);
      gemm(LNB, P(bb + 2), P(bb + 3), nullptr, BIG, M, 3 * C, C, 0);      // qkv
      attn_kernel<<<dim3(nWB, heads), 128, 0, stream>>>(BIG, P(bb + 6), ATT,
                                                        H, C, heads, shift, nWps);
      gemm(ATT, P(bb + 4), P(bb + 5), X, X, M, C, C, 0);                  // proj + res
      ln_kernel<<<M, 128, 0, stream>>>(X, P(bb + 7), P(bb + 8), LNB, C);
      gemm(LNB, P(bb + 9), P(bb + 10), nullptr, BIG, M, 4 * C, C, 1);     // fc1+gelu
      gemm(BIG, P(bb + 11), P(bb + 12), X, X, M, C, 4 * C, 0);            // fc2 + res
    }
    if (s < 3) {
      int mb = base[s] + DEPa[s] * 13;
      int Mo = M / 4;
      int tot = Mo * 4 * C;
      merge_kernel<<<(tot + 255) / 256, 256, 0, stream>>>(X, TMP, H, C);
      ln_kernel<<<Mo, 128, 0, stream>>>(TMP, P(mb + 0), P(mb + 1), LNB, 4 * C);
      gemm(LNB, P(mb + 2), nullptr, nullptr, X, Mo, 2 * C, 4 * C, 0);
    }
  }

  float* out = (float*)d_out;
  feat_kernel<<<(8 * 1024 * 49 + 255) / 256, 256, 0, stream>>>(X, out + 80);
  ln_kernel<<<8 * 49, 128, 0, stream>>>(X, P(5), P(6), LNB, 1024);
  pool_kernel<<<(8192 + 255) / 256, 256, 0, stream>>>(LNB, POOL);
  gemm(POOL, P(7), P(8),  nullptr, CLS, 8, 21841, 1024, 0);
  gemm(CLS,  P(9), P(10), nullptr, H1,  8, 1000, 21841, 0);
  gemm(H1,  P(11), P(12), nullptr, out, 8, 10, 1000, 0);
}
